// VectorQuantiser_15590731285094
// MI455X (gfx1250) — compile-verified, matrix operand order checked by reading, not run on hardware
//
#include <hip/hip_runtime.h>
#include <hip/hip_bf16.h>

typedef __bf16 bf16_t;
typedef __attribute__((ext_vector_type(16))) __bf16 v16bf;
typedef __attribute__((ext_vector_type(8)))  float  v8f;

#define K_ENT   1024
#define DIM     256
#define NTOK    65536
#define QELEMS  16777216   // 64*256*32*32

// workspace layout (bytes)
#define WS_CBB    0                           // bf16 codebook, B-fragment layout: 1024*256*2 = 512 KB
#define WS_CNORM  (K_ENT * DIM * 2)           // fp32 ||c||^2 : 4 KB
#define WS_COUNTS (WS_CNORM + K_ENT * 4)      // uint histogram : 4 KB
#define WS_PART   (WS_COUNTS + K_ENT * 4)     // fp32 per-block loss partials : 4 KB

// ---------------------------------------------------------------------------
// Repack fp32 codebook -> bf16 in WMMA B-fragment order.
// For 16-bit B (32x16): lane L holds column n = L%16, half h = L/16;
// element e = 2v+j maps to K = (v<4 ? 2v+8h : 16+2(v-4)+8h) + j.
// Flat layout: ((kt*8 + chunk)*32 + lane)*16 + e   (kt = code tile, chunk = 32-wide K slice)
// ---------------------------------------------------------------------------
__global__ void vq_prep_cb(const float* __restrict__ cb, bf16_t* __restrict__ cbB) {
    int flat  = blockIdx.x * 256 + threadIdx.x;       // 0 .. 262143
    int e     = flat & 15;
    int lane  = (flat >> 4) & 31;
    int chunk = (flat >> 9) & 7;
    int kt    = flat >> 12;
    int n = lane & 15, h = lane >> 4;
    int v = e >> 1,   j = e & 1;
    int kl = (v < 4 ? (2 * v + 8 * h) : (16 + 2 * (v - 4) + 8 * h)) + j;
    int k  = kt * 16 + n;
    int d  = chunk * 32 + kl;
    cbB[flat] = (bf16_t)cb[k * DIM + d];
}

// fp32 codebook norms + zero the histogram (must happen every launch)
__global__ void vq_prep_norm(const float* __restrict__ cb,
                             float* __restrict__ cnorm,
                             unsigned* __restrict__ counts) {
    int k = blockIdx.x;
    int t = threadIdx.x;                 // 64 threads = 2 waves
    const float4* row = (const float4*)(cb + k * DIM);
    float4 f = row[t];
    float s = f.x * f.x + f.y * f.y + f.z * f.z + f.w * f.w;
    for (int off = 16; off; off >>= 1) s += __shfl_xor(s, off);
    __shared__ float red[2];
    if ((t & 31) == 0) red[t >> 5] = s;
    __syncthreads();
    if (t == 0) { cnorm[k] = red[0] + red[1]; counts[k] = 0u; }
}

// ---------------------------------------------------------------------------
// Main fused kernel: 1024 blocks x 128 threads (4 wave32s), 64 tokens/block.
// ---------------------------------------------------------------------------
__global__ __launch_bounds__(128) void vq_main(
        const float*   __restrict__ x,        // [64,256,32,32] NCHW
        const float*   __restrict__ cb,       // [1024,256] fp32
        const bf16_t*  __restrict__ cbB_,     // B-fragment bf16 codebook
        const float*   __restrict__ cnorm,
        unsigned*      __restrict__ counts,
        float*         __restrict__ partials,
        float*         __restrict__ out) {

    __shared__ v16bf aTile[4 * 8 * 32];   // 32 KB: A fragments for 64 tokens x 256 K
    __shared__ int   idxTok[64];
    __shared__ float red[4];

    const int t       = threadIdx.x;
    const int tokbase = blockIdx.x * 64;          // 64 | 1024 -> all tokens share batch b
    const int b       = tokbase >> 10;
    const int hwbase  = tokbase & 1023;

    bf16_t* aE = (bf16_t*)aTile;

    // --- stage: coalesced fp32 load of x tile, convert to bf16 A-fragment layout ---
    for (int it = 0; it < 128; ++it) {
        int d   = it * 2 + (t >> 6);              // channel
        int tok = t & 63;                         // token within tile (contiguous in HW)
        float xv = x[((b * DIM + d) << 10) + hwbase + tok];
        int m  = tok >> 4, Ml = tok & 15;         // wave row-tile, M lane
        int chunk = d >> 5, kl = d & 31;
        int hi = kl >> 4, h = (kl >> 3) & 1, r = kl & 7;
        int v = hi * 4 + (r >> 1), j = r & 1;
        int lane = h * 16 + Ml;
        int e = 2 * v + j;
        aE[(((m << 3) + chunk) * 32 + lane) * 16 + e] = (bf16_t)xv;
    }
    __syncthreads();

    const int lane = t & 31;
    const int m    = t >> 5;       // wave id = token row-tile
    const int n    = lane & 15;    // codebook column within tile
    const v16bf* cbB = (const v16bf*)cbB_;

    // A fragments for this wave's 16 tokens live in registers (8 x v16bf = 64 VGPRs)
    v16bf aF[8];
#pragma unroll
    for (int c = 0; c < 8; ++c) aF[c] = aTile[((m << 3) + c) * 32 + lane];

    float best[8];
    int   bidx[8];
#pragma unroll
    for (int r = 0; r < 8; ++r) { best[r] = 3.4e38f; bidx[r] = 0; }

    // --- hot loop: 64 code tiles x 8 bf16 WMMAs (K = 256) ---
    for (int kt = 0; kt < 64; ++kt) {
        v8f acc = {};
#pragma unroll
        for (int c = 0; c < 8; ++c) {
            v16bf bF = cbB[((kt << 3) + c) * 32 + lane];
            acc = __builtin_amdgcn_wmma_f32_16x16x32_bf16(
                    false, aF[c], false, bF, (short)0, acc, false, false);
        }
        float cn = cnorm[(kt << 4) + n];          // fp32 ||c||^2 correction
#pragma unroll
        for (int r = 0; r < 8; ++r) {
            float dist = cn - 2.0f * acc[r];      // ||x||^2 constant per token -> dropped
            if (dist < best[r]) { best[r] = dist; bidx[r] = (kt << 4) + n; }
        }
    }

    // --- cross-lane argmin over the 16 codebook columns (first-index tie-break) ---
#pragma unroll
    for (int r = 0; r < 8; ++r) {
        float v = best[r];
        int   i = bidx[r];
        for (int off = 1; off < 16; off <<= 1) {
            float ov = __shfl_xor(v, off);
            int   oi = __shfl_xor(i, off);
            if (ov < v || (ov == v && oi < i)) { v = ov; i = oi; }
        }
        if (n == 0) {   // lanes 0 and 16 hold tokens r and r+8 of this wave's tile
            int tok = (m << 4) + r + ((lane >> 4) << 3);
            idxTok[tok] = i;
            atomicAdd(&counts[i], 1u);
            out[1 + QELEMS + 1 + tokbase + tok] = (float)i;   // idx output (as float)
        }
    }
    __syncthreads();

    // --- epilogue: gather fp32 code rows, write quantized (== STE output), MSE partial ---
    float s = 0.0f;
    for (int it = 0; it < 128; ++it) {
        int d   = it * 2 + (t >> 6);
        int tok = t & 63;
        int ci  = idxTok[tok];
        float q  = cb[ci * DIM + d];
        float xv = x[((b * DIM + d) << 10) + hwbase + tok];
        float diff = q - xv;
        s += diff * diff;
        out[1 + ((b * DIM + d) << 10) + hwbase + tok] = q;    // NCHW, coalesced
    }
    for (int off = 16; off; off >>= 1) s += __shfl_xor(s, off);
    if (lane == 0) red[m] = s;
    __syncthreads();
    if (t == 0) partials[blockIdx.x] = red[0] + red[1] + red[2] + red[3];
}

// ---------------------------------------------------------------------------
// Scalars: loss = 1.25 * mean((q-x)^2) ; perplexity = exp(-sum p log(p+1e-10))
// ---------------------------------------------------------------------------
__global__ void vq_final(const unsigned* __restrict__ counts,
                         const float* __restrict__ partials,
                         float* __restrict__ out) {
    __shared__ float rs[32], re[32];
    int t = threadIdx.x;                      // 1024 threads
    float ps  = partials[t];
    float p   = (float)counts[t] * (1.0f / 65536.0f);
    float ent = p * logf(p + 1e-10f);
    for (int off = 16; off; off >>= 1) { ps += __shfl_xor(ps, off); ent += __shfl_xor(ent, off); }
    if ((t & 31) == 0) { rs[t >> 5] = ps; re[t >> 5] = ent; }
    __syncthreads();
    if (t < 32) {
        ps  = rs[t];
        ent = re[t];
        for (int off = 16; off; off >>= 1) { ps += __shfl_xor(ps, off); ent += __shfl_xor(ent, off); }
        if (t == 0) {
            out[0]          = 1.25f * ps / 16777216.0f;   // q_loss + 0.25*e_loss, equal values
            out[1 + QELEMS] = expf(-ent);
        }
    }
}

extern "C" void kernel_launch(void* const* d_in, const int* in_sizes, int n_in,
                              void* d_out, int out_size, void* d_ws, size_t ws_size,
                              hipStream_t stream) {
    const float* x  = (const float*)d_in[0];   // [64,256,32,32] fp32
    const float* cb = (const float*)d_in[1];   // [1024,256] fp32
    float* out = (float*)d_out;

    bf16_t*   cbB     = (bf16_t*)d_ws;
    float*    cnorm   = (float*)((char*)d_ws + WS_CNORM);
    unsigned* counts  = (unsigned*)((char*)d_ws + WS_COUNTS);
    float*    partials= (float*)((char*)d_ws + WS_PART);

    vq_prep_cb  <<<1024, 256, 0, stream>>>(cb, cbB);
    vq_prep_norm<<<1024,  64, 0, stream>>>(cb, cnorm, counts);
    vq_main     <<<1024, 128, 0, stream>>>(x, cb, cbB, cnorm, counts, partials, out);
    vq_final    <<<1,   1024, 0, stream>>>(counts, partials, out);
}
